// ProbSparseMultiheadAttention_721554506367
// MI455X (gfx1250) — matrix operand order repaired and checked by reading, not compiled, not run on hardware
//
#include <hip/hip_runtime.h>
#include <hip/hip_bf16.h>
#include <math.h>

// Problem constants (from reference)
#define B_   8
#define L_   4096
#define DM_  1024
#define H_   16
#define DH_  64
#define SK_  45      // SAMPLE_K = 5*ceil(ln(4096)) = 45
#define TU_  45      // TOP_U
#define SCALE_ 0.125f // 1/sqrt(64)

typedef __attribute__((ext_vector_type(16))) __bf16 v16bf;
typedef __attribute__((ext_vector_type(8)))  __bf16 v8bf;
typedef __attribute__((ext_vector_type(8)))  float  v8f;

// round-to-nearest-even f32 -> bf16, two at a time, packed into one dword
__device__ __forceinline__ unsigned pack2bf(float a, float b) {
  union { float f; unsigned u; } xa, xb;
  xa.f = a; xb.f = b;
  const unsigned ra = (xa.u + 0x7FFFu + ((xa.u >> 16) & 1u)) >> 16;
  const unsigned rb = (xb.u + 0x7FFFu + ((xb.u >> 16) & 1u)) >> 16;
  return (rb << 16) | (ra & 0xFFFFu);
}

// ---------------------------------------------------------------------------
// One-time f32 -> bf16 conversion (4 elements / thread, vector load+store)
// ---------------------------------------------------------------------------
__global__ __launch_bounds__(256) void cvt_bf16_kernel(
    const float* __restrict__ in, unsigned* __restrict__ out, size_t n4)
{
  const size_t i = (size_t)blockIdx.x * 256 + threadIdx.x;
  if (i >= n4) return;
  const float4 f = ((const float4*)in)[i];
  uint2 o;
  o.x = pack2bf(f.x, f.y);
  o.y = pack2bf(f.z, f.w);
  ((uint2*)out)[i] = o;
}

// ---------------------------------------------------------------------------
// C[M,N] = A[M,K] @ W[N,K]^T + bias[N]    (A, W pre-converted bf16; f32 accum)
// Block: 256 threads = 8 waves, 2(M) x 4(N); wave tile 64x32 (4x2 WMMA).
// Block tile: 128 x 128. Requires M%128==0, N%128==0, K%32==0.
// Per-lane fragments use one contiguous k order (k = 16*half + s) for 16-bit A (16x32) and
// B (32x16): contiguous bf16 runs -> direct global_load_b128 into VGPRs.
// ---------------------------------------------------------------------------
__global__ __launch_bounds__(256) void gemm_bf16wmma(
    const __bf16* __restrict__ A, const __bf16* __restrict__ W,
    const float* __restrict__ bias, float* __restrict__ C,
    int M, int N, int K)
{
  const int lane = threadIdx.x & 31;
  const int wave = threadIdx.x >> 5;
  const int wm = wave & 1, wn = wave >> 1;
  const int rowBase = blockIdx.x * 128 + wm * 64;
  const int colBase = blockIdx.y * 128 + wn * 32;
  const int mrow = lane & 15;     // M (A) or N (B) index within tile
  const int half = lane >> 4;     // K-half selector

  v8f acc[4][2];
  #pragma unroll
  for (int mt = 0; mt < 4; ++mt)
    #pragma unroll
    for (int nt = 0; nt < 2; ++nt) {
      v8f z = {0.f,0.f,0.f,0.f,0.f,0.f,0.f,0.f};
      acc[mt][nt] = z;
    }

  for (int k0 = 0; k0 < K; k0 += 32) {
    v16bf af[4], bf_[2];
    // A fragment: elems 0..7  = A[row][k0 + half*16 .. +7]
    //             elems 8..15 = A[row][k0 + half*16 + 8 .. +7]
    #pragma unroll
    for (int mt = 0; mt < 4; ++mt) {
      const __bf16* ap = A + (size_t)(rowBase + mt*16 + mrow) * K + k0 + half*16;
      __builtin_prefetch(ap + 32, 0, 0);   // next k-step
      const v8bf lo = *(const v8bf*)ap;
      const v8bf hi = *(const v8bf*)(ap + 8);
      af[mt] = __builtin_shufflevector(lo, hi,
               0,1,2,3,4,5,6,7,8,9,10,11,12,13,14,15);
    }
    // B fragment (B = W^T): elems 0..15 = W[n][k0 + half*16 .. +15]
    #pragma unroll
    for (int nt = 0; nt < 2; ++nt) {
      const __bf16* wp = W + (size_t)(colBase + nt*16 + mrow) * K + k0 + half*16;
      bf_[nt] = *(const v16bf*)wp;
    }
    #pragma unroll
    for (int mt = 0; mt < 4; ++mt)
      #pragma unroll
      for (int nt = 0; nt < 2; ++nt)
        acc[mt][nt] = __builtin_amdgcn_wmma_f32_16x16x32_bf16(
            false, af[mt], false, bf_[nt],
            (short)0, acc[mt][nt], false, false);
  }

  // C/D layout: VGPR r -> M = r (lanes 0-15) / 8+r (lanes 16-31); N = lane%16
  #pragma unroll
  for (int mt = 0; mt < 4; ++mt) {
    #pragma unroll
    for (int nt = 0; nt < 2; ++nt) {
      const int col = colBase + nt*16 + mrow;
      const float bv = bias[col];
      #pragma unroll
      for (int r = 0; r < 8; ++r) {
        const int row = rowBase + mt*16 + r + half*8;
        C[(size_t)row * N + col] = acc[mt][nt][r] + bv;
      }
    }
  }
}

// ---------------------------------------------------------------------------
// m[b,h,q] = max_s(scale*q.ks_s) - mean_s(scale*q.ks_s) over SK_ sampled keys
// ---------------------------------------------------------------------------
__global__ __launch_bounds__(256) void sample_scores_kernel(
    const float* __restrict__ Q, const float* __restrict__ Kb,
    const int* __restrict__ sidx, float* __restrict__ mout)
{
  __shared__ float ks[SK_ * DH_];
  const int bh = blockIdx.x;
  const int b = bh / H_, h = bh % H_;
  for (int i = threadIdx.x; i < SK_ * DH_; i += 256) {
    const int s = i / DH_, d = i % DH_;
    const int row = sidx[s];
    ks[i] = Kb[((size_t)b * L_ + row) * DM_ + h * DH_ + d];
  }
  __syncthreads();
  const int qi = blockIdx.y * 256 + threadIdx.x;
  const float* qp = Q + ((size_t)b * L_ + qi) * DM_ + h * DH_;
  float qv[DH_];
  #pragma unroll
  for (int d = 0; d < DH_; ++d) qv[d] = qp[d];
  float mx = -INFINITY, sum = 0.f;
  for (int s = 0; s < SK_; ++s) {
    float dot = 0.f;
    #pragma unroll
    for (int d = 0; d < DH_; ++d) dot += qv[d] * ks[s * DH_ + d];
    dot *= SCALE_;
    mx = fmaxf(mx, dot);
    sum += dot;
  }
  mout[(size_t)bh * L_ + qi] = mx - sum * (1.0f / SK_);
}

// ---------------------------------------------------------------------------
// Iterated argmax top-45 per (b,h); stable tie-break to lower index
// ---------------------------------------------------------------------------
__global__ __launch_bounds__(256) void topk_kernel(
    const float* __restrict__ mbuf, int* __restrict__ tidx)
{
  __shared__ float sv[L_];
  __shared__ float rv[256];
  __shared__ int   ri[256];
  const int bh = blockIdx.x;
  for (int i = threadIdx.x; i < L_; i += 256) sv[i] = mbuf[(size_t)bh * L_ + i];
  __syncthreads();
  for (int u = 0; u < TU_; ++u) {
    float best = -INFINITY; int bi = L_;
    for (int i = threadIdx.x; i < L_; i += 256) {
      const float v = sv[i];
      if (v > best) { best = v; bi = i; }
    }
    rv[threadIdx.x] = best; ri[threadIdx.x] = bi;
    __syncthreads();
    for (int off = 128; off > 0; off >>= 1) {
      if ((int)threadIdx.x < off) {
        const float ov = rv[threadIdx.x + off]; const int oi = ri[threadIdx.x + off];
        if (ov > rv[threadIdx.x] || (ov == rv[threadIdx.x] && oi < ri[threadIdx.x])) {
          rv[threadIdx.x] = ov; ri[threadIdx.x] = oi;
        }
      }
      __syncthreads();
    }
    if (threadIdx.x == 0) {
      tidx[bh * TU_ + u] = ri[0];
      sv[ri[0]] = -INFINITY;
    }
    __syncthreads();
  }
}

// vmean[b,h,d] = mean_l V[b,l,h*64+d]
__global__ __launch_bounds__(256) void vmean_kernel(
    const float* __restrict__ V, float* __restrict__ vmean)
{
  __shared__ float red[256];
  const int bh = blockIdx.x, b = bh / H_, h = bh % H_;
  const int d = threadIdx.x & 63, part = threadIdx.x >> 6;
  float s = 0.f;
  for (int l = part; l < L_; l += 4)
    s += V[((size_t)b * L_ + l) * DM_ + h * DH_ + d];
  red[threadIdx.x] = s;
  __syncthreads();
  if (part == 0) {
    const float t = red[d] + red[64 + d] + red[128 + d] + red[192 + d];
    vmean[(size_t)bh * DH_ + d] = t * (1.0f / L_);
  }
}

// ctx[b,l,h*64+d] = vmean[b,h,d]
__global__ __launch_bounds__(256) void fill_ctx_kernel(
    const float* __restrict__ vmean, float* __restrict__ ctx)
{
  const size_t i = (size_t)blockIdx.x * 256 + threadIdx.x;
  const int c = (int)(i % DM_);
  const size_t bl = i / DM_;
  const int b = (int)(bl / L_);
  const int h = c / DH_, d = c % DH_;
  ctx[i] = vmean[((size_t)b * H_ + h) * DH_ + d];
}

// ---------------------------------------------------------------------------
// Full attention for the 45 top queries per (b,h), online softmax per wave,
// result overwrites the corresponding ctx rows.
// ---------------------------------------------------------------------------
__global__ __launch_bounds__(256) void flash_top_kernel(
    const float* __restrict__ Q, const float* __restrict__ Kb,
    const float* __restrict__ V, const int* __restrict__ tidx,
    float* __restrict__ ctx)
{
  __shared__ float qt[TU_ * DH_];
  const int bh = blockIdx.x, b = bh / H_, h = bh % H_;
  for (int i = threadIdx.x; i < TU_ * DH_; i += 256) {
    const int u = i / DH_, d = i % DH_;
    const int row = tidx[bh * TU_ + u];
    qt[i] = Q[((size_t)b * L_ + row) * DM_ + h * DH_ + d];
  }
  __syncthreads();
  const int wave = threadIdx.x >> 5, lane = threadIdx.x & 31;
  const int d0 = lane * 2, d1 = d0 + 1;
  for (int u = wave; u < TU_; u += 8) {
    float mrun = -INFINITY, lrun = 0.f, a0 = 0.f, a1 = 0.f;
    for (int c = 0; c < L_; c += 32) {
      const float* kp = Kb + ((size_t)b * L_ + c + lane) * DM_ + h * DH_;
      float s = 0.f;
      #pragma unroll
      for (int d = 0; d < DH_; ++d) s += qt[u * DH_ + d] * kp[d];
      s *= SCALE_;
      float cm = s;
      #pragma unroll
      for (int off = 16; off > 0; off >>= 1) cm = fmaxf(cm, __shfl_xor(cm, off, 32));
      const float nm = fmaxf(mrun, cm);
      const float p = __expf(s - nm);
      float ps = p;
      #pragma unroll
      for (int off = 16; off > 0; off >>= 1) ps += __shfl_xor(ps, off, 32);
      const float corr = __expf(mrun - nm);
      lrun = lrun * corr + ps;
      a0 *= corr; a1 *= corr;
      #pragma unroll 4
      for (int j = 0; j < 32; ++j) {
        const float pj = __shfl(p, j, 32);
        const float* vp = V + ((size_t)b * L_ + c + j) * DM_ + h * DH_;
        a0 += pj * vp[d0];
        a1 += pj * vp[d1];
      }
      mrun = nm;
    }
    const int row = tidx[bh * TU_ + u];
    float* op = ctx + ((size_t)b * L_ + row) * DM_ + h * DH_;
    const float inv = 1.0f / lrun;
    op[d0] = a0 * inv;
    op[d1] = a1 * inv;
  }
}

// ---------------------------------------------------------------------------
extern "C" void kernel_launch(void* const* d_in, const int* in_sizes, int n_in,
                              void* d_out, int out_size, void* d_ws, size_t ws_size,
                              hipStream_t stream)
{
  const float* x  = (const float*)d_in[0];
  const float* Wq = (const float*)d_in[1];
  const float* bq = (const float*)d_in[2];
  const float* Wk = (const float*)d_in[3];
  const float* bk = (const float*)d_in[4];
  const float* Wv = (const float*)d_in[5];
  const float* bv = (const float*)d_in[6];
  const float* Wo = (const float*)d_in[7];
  const float* bo = (const float*)d_in[8];
  const int* sidx = (const int*)d_in[9];
  float* out = (float*)d_out;

  const size_t nBig = (size_t)B_ * L_ * DM_;          // 33,554,432
  const size_t nW   = (size_t)DM_ * DM_;              // 1,048,576

  // f32 scratch
  float* Qb   = (float*)d_ws;
  float* Kb   = Qb + nBig;
  float* Vb   = Kb + nBig;
  float* CTX  = Vb + nBig;
  float* Mb   = CTX + nBig;                           // B*H*L
  float* VM   = Mb + (size_t)B_ * H_ * L_;            // B*H*DH
  // bf16 scratch (all offsets stay 32B-aligned)
  __bf16* XBF   = (__bf16*)(VM + (size_t)B_ * H_ * DH_);
  __bf16* CTXBF = XBF + nBig;
  __bf16* WQBF  = CTXBF + nBig;
  __bf16* WKBF  = WQBF + nW;
  __bf16* WVBF  = WKBF + nW;
  __bf16* WOBF  = WVBF + nW;
  int*    TIDX  = (int*)(WOBF + nW);

  const dim3 gb(256);
  const dim3 gg((B_ * L_) / 128, DM_ / 128);          // (256, 8)

  // one-time bf16 conversion of GEMM operands
  cvt_bf16_kernel<<<(unsigned)(nBig / 4 / 256), 256, 0, stream>>>(x,  (unsigned*)XBF,  nBig / 4);
  cvt_bf16_kernel<<<(unsigned)(nW   / 4 / 256), 256, 0, stream>>>(Wq, (unsigned*)WQBF, nW / 4);
  cvt_bf16_kernel<<<(unsigned)(nW   / 4 / 256), 256, 0, stream>>>(Wk, (unsigned*)WKBF, nW / 4);
  cvt_bf16_kernel<<<(unsigned)(nW   / 4 / 256), 256, 0, stream>>>(Wv, (unsigned*)WVBF, nW / 4);
  cvt_bf16_kernel<<<(unsigned)(nW   / 4 / 256), 256, 0, stream>>>(Wo, (unsigned*)WOBF, nW / 4);

  // projections (WMMA bf16, f32 accumulate)
  gemm_bf16wmma<<<gg, gb, 0, stream>>>(XBF, WQBF, bq, Qb, B_ * L_, DM_, DM_);
  gemm_bf16wmma<<<gg, gb, 0, stream>>>(XBF, WKBF, bk, Kb, B_ * L_, DM_, DM_);
  gemm_bf16wmma<<<gg, gb, 0, stream>>>(XBF, WVBF, bv, Vb, B_ * L_, DM_, DM_);

  // ProbSparse selection + sparse attention
  sample_scores_kernel<<<dim3(B_ * H_, L_ / 256), 256, 0, stream>>>(Qb, Kb, sidx, Mb);
  topk_kernel<<<B_ * H_, 256, 0, stream>>>(Mb, TIDX);
  vmean_kernel<<<B_ * H_, 256, 0, stream>>>(Vb, VM);
  fill_ctx_kernel<<<(unsigned)(nBig / 256), 256, 0, stream>>>(VM, CTX);
  flash_top_kernel<<<B_ * H_, 256, 0, stream>>>(Qb, Kb, Vb, TIDX, CTX);

  // output projection
  cvt_bf16_kernel<<<(unsigned)(nBig / 4 / 256), 256, 0, stream>>>(CTX, (unsigned*)CTXBF, nBig / 4);
  gemm_bf16wmma<<<gg, gb, 0, stream>>>(CTXBF, WOBF, bo, out, B_ * L_, DM_, DM_);
}
